// RNN_Net_34686155882861
// MI455X (gfx1250) — matrix operand order repaired and checked by reading, not compiled, not run on hardware
//
#include <hip/hip_runtime.h>
#include <hip/hip_bf16.h>

// ---------------------------------------------------------------------------
// Fused 3-layer ReLU RNN + FC head for gfx1250 (MI455X).
// One kernel: each workgroup stages ALL weights (f16, transposed) into LDS,
// then each wave (wave32) owns a 16-row batch tile and runs the full T=28
// recurrence across all 3 layers with v_wmma_f32_16x16x32_f16, keeping
// hidden states in per-wave LDS tiles. Only x is streamed from HBM.
// Final form (= round-2 structure):
//  - inline-0 WMMA accumulator init, biases hoisted to VGPRs
//  - float4 x loads with double-buffered staging (t+1 staged during t)
//  - natural scheduler order already overlaps each WMMA with the next
//    operand pair's LDS loads.
// ---------------------------------------------------------------------------

typedef __attribute__((ext_vector_type(16))) _Float16 v16h;
typedef __attribute__((ext_vector_type(8)))  _Float16 v8h;
typedef __attribute__((ext_vector_type(4)))  _Float16 v4h;
typedef __attribute__((ext_vector_type(8)))  float    v8f;

#define B_TOT 16384
#define T_LEN 28
#define I_DIM 28
#define H_DIM 128
#define C_DIM 11

#define WAVES   8
#define BLOCK   (WAVES * 32)
#define ROWS_WG (WAVES * 16)   // 128 batch rows per workgroup

// ---- LDS layout (offsets in _Float16 units unless noted) -------------------
#define OFF_WI0T 0               // [32][128]  (K padded 28->32, transposed)
#define OFF_WH0T 4096            // [128][128] transposed
#define OFF_WI1T 20480
#define OFF_WH1T 36864
#define OFF_WI2T 53248
#define OFF_WH2T 69632
#define OFF_FCT  86016           // [128][16]  (N padded 11->16, transposed)
#define OFF_H0   88064           // 8 waves * 16 rows * 128
#define OFF_H1   104448
#define OFF_H2   120832
#define OFF_XS   137216          // [2 bufs][8 waves][16 rows][32]  (K pad 28->32)
#define XS_BUF   (WAVES * 16 * 32)              // 4096 halfs per buffer
#define HALF_END (OFF_XS + 2 * XS_BUF)          // 145408
#define OFF_BIAS_BYTES (HALF_END * 2)           // 3*128 + 16 floats
#define LDS_BYTES (OFF_BIAS_BYTES + 400 * 4)    // = 292416  (< 320KB)

// ---------------------------------------------------------------------------
// Operand loaders (layouts per CDNA5 ISA 7.12.2).
// A (16x32, f16): lanes 0-15 & 16-31 both hold rows M=0..15; lane half selects
//   K sub-blocks {0..7,16..23} vs {8..15,24..31}. Two 16B LDS reads.
// B (32x16, f16): lane holds column N = lane%16; elements span K in the same
//   order as A ({0..7,16..23} vs {8..15,24..31} by lane half). Weights are
//   staged transposed [K][N], so each element is a strided scalar LDS read.
// ---------------------------------------------------------------------------
__device__ __forceinline__ v16h load_a(const _Float16* A, int stride, int kc, int lane) {
    const int row = lane & 15;
    const int sel = (lane & 16) ? 8 : 0;
    const _Float16* p = A + row * stride + kc * 32 + sel;
    v8h lo = *(const v8h*)(p);
    v8h hi = *(const v8h*)(p + 16);
    return __builtin_shufflevector(lo, hi, 0,1,2,3,4,5,6,7,8,9,10,11,12,13,14,15);
}

__device__ __forceinline__ v16h load_b(const _Float16* wT, int rowStride, int kc, int nt, int lane) {
    const int k0 = kc * 32 + ((lane & 16) ? 8 : 0);
    const _Float16* p = wT + k0 * rowStride + nt * 16 + (lane & 15);
    v16h r;
#pragma unroll
    for (int i = 0; i < 8; ++i) {
        r[i]     = p[i * rowStride];
        r[8 + i] = p[(16 + i) * rowStride];
    }
    return r;
}

__device__ __forceinline__ v8f wmma_f16(v16h a, v16h b, v8f c) {
    return __builtin_amdgcn_wmma_f32_16x16x32_f16(false, a, false, b, (short)0, c, false, false);
}

// One RNN layer step for one wave's 16-row tile:
//   H_out = relu( A1 @ B1 + A2 @ B2 + bias ), all 128 outputs.
// Accumulators start at zero (lowers to WMMA inline-0 SRC2); per-lane bias
// scalars live in VGPRs (bReg) and are folded into the ReLU tail.
template <int NC1>
__device__ __forceinline__ void rnn_layer_step(
    const _Float16* A1, int s1, const _Float16* B1,
    const _Float16* A2, const _Float16* B2,
    const float* bReg, _Float16* Hout, int lane)
{
    const int nlo = lane & 15;
    const int mhi = (lane & 16) ? 8 : 0;

    v8f acc[8] = {};   // zero-init -> first WMMA uses inline 0 as C operand

#pragma unroll
    for (int kc = 0; kc < NC1; ++kc) {
        v16h a = load_a(A1, s1, kc, lane);
#pragma unroll
        for (int nt = 0; nt < 8; ++nt) {
            v16h b = load_b(B1, 128, kc, nt, lane);
            acc[nt] = wmma_f16(a, b, acc[nt]);
        }
    }
#pragma unroll
    for (int kc = 0; kc < 4; ++kc) {
        v16h a = load_a(A2, 128, kc, lane);
#pragma unroll
        for (int nt = 0; nt < 8; ++nt) {
            v16h b = load_b(B2, 128, kc, nt, lane);
            acc[nt] = wmma_f16(a, b, acc[nt]);
        }
    }

    // bias + ReLU + f16 convert; D layout: VGPR v -> M = v + mhi, N = nlo+16*nt.
#pragma unroll
    for (int nt = 0; nt < 8; ++nt) {
#pragma unroll
        for (int v = 0; v < 8; ++v) {
            float f = acc[nt][v] + bReg[nt];
            f = f > 0.0f ? f : 0.0f;
            Hout[(v + mhi) * 128 + nt * 16 + nlo] = (_Float16)f;
        }
    }
}

// Stage one timestep's x tile (16 rows x 28 f32 -> f16, K padded to 32).
// Rows are 16B aligned (784 and 28 are multiples of 4), so use float4 loads
// and 8-byte packed LDS stores: 112 float4 segments over 32 lanes.
__device__ __forceinline__ void stage_x(const float* __restrict__ x,
                                        _Float16* xsbuf, int row0, int t, int lane)
{
    for (int i = lane; i < 16 * 7; i += 32) {
        const int r = i / 7, q = i - r * 7;
        const float4 v = *(const float4*)&x[(size_t)(row0 + r) * (T_LEN * I_DIM) + t * I_DIM + q * 4];
        v4h hv;
        hv[0] = (_Float16)v.x; hv[1] = (_Float16)v.y;
        hv[2] = (_Float16)v.z; hv[3] = (_Float16)v.w;
        *(v4h*)&xsbuf[r * 32 + q * 4] = hv;
    }
}

__device__ __forceinline__ void stage_sq_T(_Float16* dst, const float* src, int tid) {
    // dst[k*128 + n] = src[n*128 + k]  (transpose to [K][N])
    for (int i = tid; i < 128 * 128; i += BLOCK) {
        const int k = i >> 7, n = i & 127;
        dst[i] = (_Float16)src[n * 128 + k];
    }
}

__global__ void __launch_bounds__(BLOCK, 1)
rnn3_fused_wmma(const float* __restrict__ x,
                const float* __restrict__ wi0, const float* __restrict__ wh0,
                const float* __restrict__ bi0, const float* __restrict__ bh0,
                const float* __restrict__ wi1, const float* __restrict__ wh1,
                const float* __restrict__ bi1, const float* __restrict__ bh1,
                const float* __restrict__ wi2, const float* __restrict__ wh2,
                const float* __restrict__ bi2, const float* __restrict__ bh2,
                const float* __restrict__ fcw, const float* __restrict__ fcb,
                float* __restrict__ out)
{
    extern __shared__ _Float16 smem[];
    const int tid = threadIdx.x;

    // ---- Stage all weights into LDS (f16, transposed [K][N]) --------------
    stage_sq_T(smem + OFF_WH0T, wh0, tid);
    stage_sq_T(smem + OFF_WI1T, wi1, tid);
    stage_sq_T(smem + OFF_WH1T, wh1, tid);
    stage_sq_T(smem + OFF_WI2T, wi2, tid);
    stage_sq_T(smem + OFF_WH2T, wh2, tid);
    for (int i = tid; i < 32 * 128; i += BLOCK) {          // wi0: [28->32][128]
        const int k = i >> 7, n = i & 127;
        smem[OFF_WI0T + i] = (k < I_DIM) ? (_Float16)wi0[n * I_DIM + k] : (_Float16)0.0f;
    }
    for (int i = tid; i < 128 * 16; i += BLOCK) {          // fc: [128][11->16]
        const int k = i >> 4, n = i & 15;
        smem[OFF_FCT + i] = (n < C_DIM) ? (_Float16)fcw[n * 128 + k] : (_Float16)0.0f;
    }
    float* bias = (float*)((char*)smem + OFF_BIAS_BYTES);  // b0,b1,b2,fcb
    if (tid < 128) {
        bias[tid]       = bi0[tid] + bh0[tid];
        bias[128 + tid] = bi1[tid] + bh1[tid];
        bias[256 + tid] = bi2[tid] + bh2[tid];
    }
    if (tid < 16) bias[384 + tid] = (tid < C_DIM) ? fcb[tid] : 0.0f;
    // zero h0/h1/h2 (t=0 state) and x staging pad columns
    for (int i = tid; i < 3 * WAVES * 16 * 128; i += BLOCK) smem[OFF_H0 + i] = (_Float16)0.0f;
    for (int i = tid; i < 2 * XS_BUF; i += BLOCK)           smem[OFF_XS + i] = (_Float16)0.0f;
    __syncthreads();

    // ---- Per-wave tiles ----------------------------------------------------
    const int wave = tid >> 5;
    const int lane = tid & 31;
    const int nlo  = lane & 15;
    const int row0 = blockIdx.x * ROWS_WG + wave * 16;

    _Float16* h0  = smem + OFF_H0 + wave * (16 * 128);
    _Float16* h1  = smem + OFF_H1 + wave * (16 * 128);
    _Float16* h2  = smem + OFF_H2 + wave * (16 * 128);
    _Float16* xsA = smem + OFF_XS + wave * (16 * 32);            // buffer 0
    _Float16* xsB = smem + OFF_XS + XS_BUF + wave * (16 * 32);   // buffer 1

    const _Float16* wi0T = smem + OFF_WI0T;
    const _Float16* wh0T = smem + OFF_WH0T;
    const _Float16* wi1T = smem + OFF_WI1T;
    const _Float16* wh1T = smem + OFF_WH1T;
    const _Float16* wi2T = smem + OFF_WI2T;
    const _Float16* wh2T = smem + OFF_WH2T;

    // Hoist per-lane bias scalars into VGPRs (t-invariant; compiler cannot
    // hoist LDS loads itself because h-tile stores alias the LDS region).
    float bR0[8], bR1[8], bR2[8];
#pragma unroll
    for (int nt = 0; nt < 8; ++nt) {
        bR0[nt] = bias[      nt * 16 + nlo];
        bR1[nt] = bias[128 + nt * 16 + nlo];
        bR2[nt] = bias[256 + nt * 16 + nlo];
    }
    const float fcbv = bias[384 + nlo];

    // Pre-stage t=0, then stage t+1 while computing t (double buffer): the
    // global-load latency hides behind ~168 WMMAs of the previous timestep.
    stage_x(x, xsA, row0, 0, lane);

    for (int t = 0; t < T_LEN; ++t) {
        _Float16* xsCur = (t & 1) ? xsB : xsA;
        if (t + 1 < T_LEN) {
            stage_x(x, (t & 1) ? xsA : xsB, row0, t + 1, lane);
            __builtin_prefetch(&x[(size_t)(row0 + nlo) * (T_LEN * I_DIM) + (t + 2) * I_DIM], 0, 0);
        }

        // layer 0: pre = x@Wih0^T, rec = h0@Whh0^T      (LDS ops in-order per
        // wave guarantee h0 reads see timestep t-1 before the tail writes)
        rnn_layer_step<1>(xsCur, 32, wi0T, h0, wh0T, bR0, h0, lane);
        // layer 1: input = fresh h0, state = h1
        rnn_layer_step<4>(h0, 128, wi1T, h1, wh1T, bR1, h1, lane);
        // layer 2
        rnn_layer_step<4>(h1, 128, wi2T, h2, wh2T, bR2, h2, lane);
    }

    // ---- FC head: out[16,11] = h2_last @ fcw^T + fcb ----------------------
    {
        const int mhi = (lane & 16) ? 8 : 0;
        v8f acc = {};
#pragma unroll
        for (int kc = 0; kc < 4; ++kc) {
            v16h a = load_a(h2, 128, kc, lane);
            v16h b = load_b(smem + OFF_FCT, 16, kc, 0, lane);
            acc = wmma_f16(a, b, acc);
        }
        if (nlo < C_DIM) {
#pragma unroll
            for (int v = 0; v < 8; ++v)
                out[(size_t)(row0 + v + mhi) * C_DIM + nlo] = acc[v] + fcbv;
        }
    }
}

extern "C" void kernel_launch(void* const* d_in, const int* in_sizes, int n_in,
                              void* d_out, int out_size, void* d_ws, size_t ws_size,
                              hipStream_t stream) {
    (void)in_sizes; (void)n_in; (void)out_size; (void)d_ws; (void)ws_size;
    const float* x   = (const float*)d_in[0];
    const float* wi0 = (const float*)d_in[1];
    const float* wh0 = (const float*)d_in[2];
    const float* bi0 = (const float*)d_in[3];
    const float* bh0 = (const float*)d_in[4];
    const float* wi1 = (const float*)d_in[5];
    const float* wh1 = (const float*)d_in[6];
    const float* bi1 = (const float*)d_in[7];
    const float* bh1 = (const float*)d_in[8];
    const float* wi2 = (const float*)d_in[9];
    const float* wh2 = (const float*)d_in[10];
    const float* bi2 = (const float*)d_in[11];
    const float* bh2 = (const float*)d_in[12];
    const float* fcw = (const float*)d_in[13];
    const float* fcb = (const float*)d_in[14];

    dim3 grid(B_TOT / ROWS_WG);   // 128 workgroups
    dim3 block(BLOCK);            // 256 threads = 8 wave32
    rnn3_fused_wmma<<<grid, block, LDS_BYTES, stream>>>(
        x, wi0, wh0, bi0, bh0, wi1, wh1, bi1, bh1, wi2, wh2, bi2, bh2,
        fcw, fcb, (float*)d_out);
}
